// BitLinear_63204738728231
// MI455X (gfx1250) — 8-bit matrix operand order repaired and checked by reading, compiled, not run on hardware
//
#include <hip/hip_runtime.h>

typedef int v8i __attribute__((ext_vector_type(8)));

#define EPS 1e-8f

__global__ __launch_bounds__(256) void wabs_partial(const float* __restrict__ w,
                                                    float* __restrict__ partial) {
    __shared__ float sdata[256];
    int t = threadIdx.x;
    int gid = blockIdx.x * 256 + t;
    float s = 0.f;
    for (int i = gid; i < 1024 * 1024; i += 256 * 256) s += fabsf(w[i]);
    sdata[t] = s;
    __syncthreads();
    for (int off = 128; off > 0; off >>= 1) {
        if (t < off) sdata[t] += sdata[t + off];
        __syncthreads();
    }
    if (t == 0) partial[blockIdx.x] = sdata[0];
}

__global__ __launch_bounds__(256) void wabs_final(const float* __restrict__ partial,
                                                  float* __restrict__ wscale) {
    __shared__ float sdata[256];
    int t = threadIdx.x;
    sdata[t] = partial[t];
    __syncthreads();
    for (int off = 128; off > 0; off >>= 1) {
        if (t < off) sdata[t] += sdata[t + off];
        __syncthreads();
    }
    if (t == 0) wscale[0] = sdata[0] / (1024.f * 1024.f) + EPS;
}

__global__ __launch_bounds__(256) void pack_weight(const float* __restrict__ w,
                                                   const float* __restrict__ wscale_p,
                                                   int* __restrict__ wpack) {
    int ntile = blockIdx.x;
    int t = threadIdx.x;
    int lane = t >> 3;
    int v = t & 7;
    float ws = wscale_p[0];
    int n = ntile * 16 + (lane & 15);
    int kh = lane >> 4;
    const float* wr = w + (size_t)n * 1024;
    for (int kb = 0; kb < 16; ++kb) {
        int kbase = kb * 64 + ((v & 4) ? 32 : 0) + kh * 16 + (v & 3) * 4;
        unsigned d = 0;
        #pragma unroll
        for (int b = 0; b < 4; ++b) {
            float u = wr[kbase + b] / ws;
            float r = rintf(u);
            r = fminf(fmaxf(r, -1.f), 1.f);
            int q = (int)r;
            d |= ((unsigned)(q & 0xFF)) << (8 * b);
        }
        wpack[((size_t)ntile * 16 + kb) * 256 + t] = (int)d;
    }
}

__global__ __launch_bounds__(256) void quant_act(const float* __restrict__ x,
                                                 const float* __restrict__ wscale_p,
                                                 int* __restrict__ apack,
                                                 float* __restrict__ rowscale) {
    __shared__ signed char sq[16][1024];
    int mtile = blockIdx.x;
    int t = threadIdx.x;
    int wave = t >> 5;
    int lane = t & 31;
    float ws = wscale_p[0];

    for (int rr = 0; rr < 2; ++rr) {
        int rlocal = wave * 2 + rr;
        int m = mtile * 16 + rlocal;
        const float* xr = x + (size_t)m * 1024;
        float vals[32];
        float amax = 0.f;
        #pragma unroll
        for (int i = 0; i < 32; ++i) {
            float vv = xr[i * 32 + lane];
            vals[i] = vv;
            amax = fmaxf(amax, fabsf(vv));
        }
        #pragma unroll
        for (int off = 16; off > 0; off >>= 1)
            amax = fmaxf(amax, __shfl_xor(amax, off, 32));
        float ascale = amax + EPS;
        if (lane == 0) rowscale[m] = ws * ascale / 127.0f;
        #pragma unroll
        for (int i = 0; i < 32; ++i) {
            float z = vals[i] / ascale * 127.0f;
            z = fminf(fmaxf(z, -128.f), 127.f);
            int q = (int)rintf(z);
            sq[rlocal][i * 32 + lane] = (signed char)q;
        }
    }
    __syncthreads();

    int lane8 = t >> 3;
    int v = t & 7;
    int mloc = lane8 & 15;
    int kh = lane8 >> 4;
    for (int kb = 0; kb < 16; ++kb) {
        int kbase = kb * 64 + ((v & 4) ? 32 : 0) + kh * 16 + (v & 3) * 4;
        unsigned d = 0;
        #pragma unroll
        for (int b = 0; b < 4; ++b)
            d |= ((unsigned)(unsigned char)sq[mloc][kbase + b]) << (8 * b);
        apack[((size_t)mtile * 16 + kb) * 256 + t] = (int)d;
    }
}

__global__ __launch_bounds__(256) void bitgemm(const int* __restrict__ apack,
                                               const int* __restrict__ wpack,
                                               const float* __restrict__ rowscale,
                                               float* __restrict__ out) {
    int t = threadIdx.x;
    int wave = t >> 5;
    int lane = t & 31;
    int gw = blockIdx.x * 8 + wave;
    int nblk = gw & 15;
    int mquad = gw >> 4;

    const v8i* Ab = (const v8i*)apack + (size_t)(mquad * 4) * 16 * 32 + lane;
    const v8i* Bb = (const v8i*)wpack + (size_t)(nblk * 4) * 16 * 32 + lane;

    v8i acc[4][4];
    #pragma unroll
    for (int mi = 0; mi < 4; ++mi)
        #pragma unroll
        for (int ni = 0; ni < 4; ++ni)
            acc[mi][ni] = (v8i){};

    v8i aBuf[2][4], bBuf[2][4];
    #pragma unroll
    for (int i = 0; i < 4; ++i) {
        aBuf[0][i] = Ab[(i * 16 + 0) * 32];
        bBuf[0][i] = Bb[(i * 16 + 0) * 32];
    }

    #pragma unroll
    for (int kb = 0; kb < 16; ++kb) {
        const int cur = kb & 1;
        const int nxt = cur ^ 1;
        if (kb < 15) {
            #pragma unroll
            for (int i = 0; i < 4; ++i) {
                aBuf[nxt][i] = Ab[(i * 16 + kb + 1) * 32];
                bBuf[nxt][i] = Bb[(i * 16 + kb + 1) * 32];
            }
        }
        #pragma unroll
        for (int mi = 0; mi < 4; ++mi)
            #pragma unroll
            for (int ni = 0; ni < 4; ++ni)
                acc[mi][ni] = __builtin_amdgcn_wmma_i32_16x16x64_iu8(
                    true, aBuf[cur][mi], true, bBuf[cur][ni], acc[mi][ni],
                    false, false);
    }

    int half = lane >> 4;
    int ncol = lane & 15;
    #pragma unroll
    for (int mi = 0; mi < 4; ++mi) {
        int mbase = (mquad * 4 + mi) * 16 + half * 8;
        const float* rs = rowscale + mbase;
        float* ob = out + (size_t)mbase * 1024 + nblk * 64 + ncol;
        #pragma unroll
        for (int r = 0; r < 8; ++r) {
            float s = rs[r];
            float* orow = ob + (size_t)r * 1024;
            #pragma unroll
            for (int ni = 0; ni < 4; ++ni)
                orow[ni * 16] = (float)acc[mi][ni][r] * s;
        }
    }
}

extern "C" void kernel_launch(void* const* d_in, const int* in_sizes, int n_in,
                              void* d_out, int out_size, void* d_ws, size_t ws_size,
                              hipStream_t stream) {
    (void)in_sizes; (void)n_in; (void)out_size; (void)ws_size;
    const float* x = (const float*)d_in[0];
    const float* w = (const float*)d_in[1];
    float* out = (float*)d_out;

    char* ws = (char*)d_ws;
    float* wscale   = (float*)(ws + 0);
    float* partial  = (float*)(ws + 256);
    int*   wpack    = (int*)  (ws + 4096);
    float* rowscale = (float*)(ws + 4096 + (1 << 20));
    int*   apack    = (int*)  (ws + 4096 + (1 << 20) + 32768 * 4);

    wabs_partial<<<256,  256, 0, stream>>>(w, partial);
    wabs_final  <<<1,    256, 0, stream>>>(partial, wscale);
    pack_weight <<<64,   256, 0, stream>>>(w, wscale, wpack);
    quant_act   <<<2048, 256, 0, stream>>>(x, wscale, apack, rowscale);
    bitgemm     <<<1024, 256, 0, stream>>>(apack, wpack, rowscale, out);
}
